// PointTransformerBlock_68281390072567
// MI455X (gfx1250) — matrix operand order repaired and checked by reading, not compiled, not run on hardware
//
// Point Transformer block, fused for MI455X (gfx1250, wave32, WMMA).
//
// Round-2 changes: operands are pre-swizzled into the WMMA register image.
//  - Weights are converted f32->f16 *and* laid out fragment-major, so a B
//    fragment is one aligned 32B global load per lane (2x global_load_b128,
//    coalesced, L2-resident).
//  - A tiles are written by their producers directly in fragment layout in
//    LDS, so an A fragment is one aligned 32B LDS load (2x ds_load_b128).
//  - block GEMM K-loop has no LDS staging and no barriers.
//
// Pipeline: cvt(swizzle) -> ln1 -> knn -> qkv[WMMA] -> fused attn[3x WMMA]
//           -> fc_out+norm2[WMMA] -> channel attn -> skip+norm3[WMMA].
// ~106 GFLOP f16 WMMA vs ~60 MB HBM traffic: compute-bound by design.

#include <hip/hip_runtime.h>
#include <math.h>

typedef __attribute__((ext_vector_type(16))) _Float16 v16h;
typedef __attribute__((ext_vector_type(8)))  float    v8f;
typedef _Float16 h16;

#define BB   2
#define NN   4096
#define CINC 128
#define COC  256
#define KNB  32
#define NPTS (BB * NN)
#define LNEPS 1e-5f

// A fragment buffer: [m-tile][k-slab] blocks of 32 lanes x 16 halves (512 h).
// Max Kc = 256 -> 8 k-slabs -> 2*8*512 halves = 16 KB.
#define FRAG_H 512

// ---------------------------------------------------------------------------
// A-operand scatter: place value for logical (row m, col k) at the (lane,
// element) demanded by the ISA 16x32 f16 A layout:
//   lanes 0-15 : M=lane,    K in {0..7, 16..23}
//   lanes 16-31: M=lane-16, K in {8..15, 24..31}
//   element j = 2t+s where k_local = (t&4 ? 16:0) + (t&3)*2 + s
// ---------------------------------------------------------------------------
__device__ __forceinline__ void store_a_elem(h16* Af, int r, int c, int Kc32, h16 v) {
  const int mt = r >> 4, mrow = r & 15;
  const int ks = c >> 5, kl = c & 31;
  const int kb = (kl & 8) ? 8 : 0;          // hi-lane group?
  const int lane = mrow + (kb ? 16 : 0);
  const int klb = kl - kb;                  // 0..7 or 16..23
  const int j = 2 * (((klb & 16) ? 4 : 0) + ((klb >> 1) & 3)) + (klb & 1);
  Af[((mt * Kc32 + ks) << 9) + (lane << 4) + j] = v;
}

__device__ __forceinline__ v16h load_a_frag(const h16* Af, int mt, int ks, int Kc32) {
  const int lane = threadIdx.x & 31;
  return *(const v16h*)(Af + ((mt * Kc32 + ks) << 9) + (lane << 4));  // 32B aligned
}

// B fragment (pre-swizzled in global): frag index = ks*16 + nt, 512 halves each.
__device__ __forceinline__ v16h load_b_frag(const h16* Wg, int ks, int nt) {
  const int lane = threadIdx.x & 31;
  return *(const v16h*)(Wg + ((((ks << 4) + nt) << 9) + (lane << 4)));
}

// ---------------------------------------------------------------------------
// Block GEMM: D[32][256] = A[32][Kc] (LDS frag image) * W[Kc][256] (global,
// frag image) + bias.  256 threads = 8 waves; wave owns 4 of 32 16x16 tiles.
// No LDS staging, no barriers in the K loop.
// Callers: __syncthreads() separates Af writes from this call, and the call
// from reads of D.
// ---------------------------------------------------------------------------
__device__ void block_gemm_32(const h16* __restrict__ Af, const h16* __restrict__ Wg,
                              const float* __restrict__ bias,
                              float* D, int ldD, int row0, int Kc32) {
  __syncthreads();  // Af produced by all threads is visible; prior D reads done
  const int wave = threadIdx.x >> 5;
  const int lane = threadIdx.x & 31;
  const int mt = wave & 1;
  const int ng = wave >> 1;  // 0..3
  v8f acc[4] = {};
  for (int ks = 0; ks < Kc32; ++ks) {
    if (ks + 1 < Kc32)
      __builtin_prefetch(Wg + (((ks + 1) << 4) << 9), 0, 0);
    const v16h a = load_a_frag(Af, mt, ks, Kc32);
#pragma unroll
    for (int t = 0; t < 4; ++t) {
      const v16h b = load_b_frag(Wg, ks, ng + 4 * t);
      acc[t] = __builtin_amdgcn_wmma_f32_16x16x32_f16(
          false, a, false, b, (short)0, acc[t], false, false);
    }
  }
  // C/D image: v8f elem r -> row = 16*mt + 8*lane[4] + r, col = n0 + (lane&15)
#pragma unroll
  for (int t = 0; t < 4; ++t) {
    const int col = (ng + 4 * t) * 16 + (lane & 15);
    const float bv = bias ? bias[col] : 0.0f;
#pragma unroll
    for (int r = 0; r < 8; ++r) {
      const int row = mt * 16 + ((lane & 16) ? 8 : 0) + r;
      D[(size_t)(row0 + row) * ldD + col] = acc[t][r] + bv;
    }
  }
}

// ---------------------------------------------------------------------------
// 1. Weight convert + swizzle into B fragment image.
//    out[i]: j=i&15, lane=(i>>4)&31, nt=(i>>9)&15, ks=i>>13
//    k = ks*32 + (lane&16 ? 8:0) + (j&7) + (j&8 ? 16:0) ; col = nt*16 + (lane&15)
// ---------------------------------------------------------------------------
__global__ void cvt_swz_b(const float* __restrict__ s, h16* __restrict__ d, int n) {
  const int i = blockIdx.x * blockDim.x + threadIdx.x;
  if (i >= n) return;
  const int j = i & 15, lane = (i >> 4) & 31, nt = (i >> 9) & 15, ks = i >> 13;
  const int k = ks * 32 + ((lane & 16) ? 8 : 0) + (j & 7) + ((j & 8) ? 16 : 0);
  const int col = nt * 16 + (lane & 15);
  d[i] = (h16)s[(size_t)k * COC + col];
}

// ---------------------------------------------------------------------------
// 2. norm1 + f16 copies. Block = 128 threads = 1 row.
// ---------------------------------------------------------------------------
__global__ void ln1_kernel(const float* __restrict__ feat, const float* __restrict__ g,
                           const float* __restrict__ bta, h16* __restrict__ fnh,
                           h16* __restrict__ fth) {
  __shared__ float red[CINC];
  const int p = blockIdx.x, c = threadIdx.x;
  const float v = feat[(size_t)p * CINC + c];
  fth[(size_t)p * CINC + c] = (h16)v;
  red[c] = v; __syncthreads();
  for (int s = CINC / 2; s > 0; s >>= 1) { if (c < s) red[c] += red[c + s]; __syncthreads(); }
  const float mu = red[0] * (1.0f / CINC); __syncthreads();
  red[c] = (v - mu) * (v - mu); __syncthreads();
  for (int s = CINC / 2; s > 0; s >>= 1) { if (c < s) red[c] += red[c + s]; __syncthreads(); }
  const float rstd = rsqrtf(red[0] * (1.0f / CINC) + LNEPS);
  fnh[(size_t)p * CINC + c] = (h16)((v - mu) * rstd * g[c] + bta[c]);
}

// ---------------------------------------------------------------------------
// 3. KNN: block per query point; 4096 distances in LDS; 32 masked argmaxes.
// ---------------------------------------------------------------------------
__global__ void knn_kernel(const float* __restrict__ xyz, int* __restrict__ idxo) {
  __shared__ float dist[NN];
  __shared__ float rv[256];
  __shared__ int   ri[256];
  const int p = blockIdx.x, b = p / NN, tid = threadIdx.x;
  const float qx = xyz[(size_t)p * 3], qy = xyz[(size_t)p * 3 + 1], qz = xyz[(size_t)p * 3 + 2];
  const float* base = xyz + (size_t)b * NN * 3;
  for (int i = tid; i < NN; i += 256) {
    const float dx = qx - base[i * 3], dy = qy - base[i * 3 + 1], dz = qz - base[i * 3 + 2];
    dist[i] = -(dx * dx + dy * dy + dz * dz);
  }
  __syncthreads();
  for (int kk = 0; kk < KNB; ++kk) {
    float bv = -3.4e38f; int bi = 0;
    for (int i = tid; i < NN; i += 256)
      if (dist[i] > bv) { bv = dist[i]; bi = i; }
    rv[tid] = bv; ri[tid] = bi; __syncthreads();
    for (int s = 128; s > 0; s >>= 1) {
      if (tid < s) {
        const float ov = rv[tid + s]; const int oi = ri[tid + s];
        if (ov > rv[tid] || (ov == rv[tid] && oi < ri[tid])) { rv[tid] = ov; ri[tid] = oi; }
      }
      __syncthreads();
    }
    if (tid == 0) { idxo[(size_t)p * KNB + kk] = ri[0]; dist[ri[0]] = -3.4e38f; }
    __syncthreads();
  }
}

// ---------------------------------------------------------------------------
// 4. q/k/v projections: block per 32 rows, 3 WMMA GEMMs reusing the A image.
// ---------------------------------------------------------------------------
struct QKVArgs {
  const h16 *fnh, *Wq, *Wk, *Wv;
  const float *bq, *bk, *bv;
  float *q, *k, *v;
};
__global__ void __launch_bounds__(256) qkv_kernel(QKVArgs a) {
  __shared__ h16 Af[2 * 4 * FRAG_H];  // Kc = 128 -> 4 k-slabs
  const int row0 = blockIdx.x * 32;
  for (int i = threadIdx.x; i < 32 * CINC; i += 256) {
    const int r = i >> 7, c = i & 127;
    store_a_elem(Af, r, c, 4, a.fnh[(size_t)(row0 + r) * CINC + c]);
  }
  block_gemm_32(Af, a.Wq, a.bq, a.q, COC, row0, 4);
  block_gemm_32(Af, a.Wk, a.bk, a.k, COC, row0, 4);
  block_gemm_32(Af, a.Wv, a.bv, a.v, COC, row0, 4);
}

// ---------------------------------------------------------------------------
// 5. Fused per-point attention: 3 WMMA GEMMs + 2 LNs + softmax + weighted sum
//    entirely in LDS (~86 KB / workgroup; WGP has 320 KB).
// ---------------------------------------------------------------------------
#define DSTR 260   // f32 D-tile stride (padded)
struct AttnArgs {
  const float* xyz; const int* idx;
  const float *q, *k, *v;
  const float *W1, *b1, *dg, *db;               // delta1 (3x256 f32), delta_ln
  const h16* W2h; const float* b2;              // delta2 (swizzled f16)
  const h16* Wg1h; const float *bg1, *gg, *gb;  // gamma1, gamma_ln
  const h16* Wg2h; const float* bg2;            // gamma2
  float* out;                                   // [NPTS, 256]
};
__global__ void __launch_bounds__(256) attn_kernel(AttnArgs a) {
  __shared__ h16   Af[2 * 8 * FRAG_H];          // Kc = 256 -> 8 k-slabs
  __shared__ float pe[32 * DSTR];
  __shared__ float sb[32 * DSTR];
  __shared__ float qrow[COC];
  __shared__ float pd[32][3];
  __shared__ int   nb[32];
  __shared__ float red1[32][8], red2[32][8];

  const int p = blockIdx.x, b = p / NN, tid = threadIdx.x;
  if (tid < 32) {
    const int j = a.idx[(size_t)p * KNB + tid];
    nb[tid] = j;
    const size_t g = (size_t)(b * NN + j) * 3, q3 = (size_t)p * 3;
    pd[tid][0] = a.xyz[q3]     - a.xyz[g];
    pd[tid][1] = a.xyz[q3 + 1] - a.xyz[g + 1];
    pd[tid][2] = a.xyz[q3 + 2] - a.xyz[g + 2];
  }
  qrow[tid] = a.q[(size_t)p * COC + tid];
  __syncthreads();

  const int r = tid >> 3, li = tid & 7;  // row 0..31, 8 lanes x 32 cols per row

  // ---- delta1 (3->256) + LN + ReLU -> A image ---------------------------
  {
    float vb[32], s = 0.f, s2 = 0.f;
    const float x0 = pd[r][0], x1 = pd[r][1], x2 = pd[r][2];
#pragma unroll
    for (int jj = 0; jj < 32; ++jj) {
      const int c = li * 32 + jj;
      const float v = a.b1[c] + x0 * a.W1[c] + x1 * a.W1[COC + c] + x2 * a.W1[2 * COC + c];
      vb[jj] = v; s += v; s2 += v * v;
    }
    red1[r][li] = s; red2[r][li] = s2; __syncthreads();
    float mu = 0.f, m2 = 0.f;
#pragma unroll
    for (int i = 0; i < 8; ++i) { mu += red1[r][i]; m2 += red2[r][i]; }
    mu *= (1.f / COC);
    const float rstd = rsqrtf(m2 * (1.f / COC) - mu * mu + LNEPS);
#pragma unroll
    for (int jj = 0; jj < 32; ++jj) {
      const int c = li * 32 + jj;
      const float h = (vb[jj] - mu) * rstd * a.dg[c] + a.db[c];
      store_a_elem(Af, r, c, 8, (h16)fmaxf(h, 0.f));
    }
  }

  // ---- pe = h1 @ delta2 + b2 --------------------------------------------
  block_gemm_32(Af, a.W2h, a.b2, pe, DSTR, 0, 8);
  __syncthreads();

  // ---- attn_input = q - k_gather + pe -> A image ------------------------
  {
    const float* krow = a.k + (size_t)(b * NN + nb[r]) * COC;
#pragma unroll
    for (int jj = 0; jj < 32; ++jj) {
      const int c = li * 32 + jj;
      store_a_elem(Af, r, c, 8, (h16)(qrow[c] - krow[c] + pe[(size_t)r * DSTR + c]));
    }
  }
  block_gemm_32(Af, a.Wg1h, a.bg1, sb, DSTR, 0, 8);
  __syncthreads();

  // ---- gamma_ln + ReLU -> A image ---------------------------------------
  {
    float s = 0.f, s2 = 0.f;
#pragma unroll
    for (int jj = 0; jj < 32; ++jj) {
      const float v = sb[(size_t)r * DSTR + li * 32 + jj]; s += v; s2 += v * v;
    }
    red1[r][li] = s; red2[r][li] = s2; __syncthreads();
    float mu = 0.f, m2 = 0.f;
#pragma unroll
    for (int i = 0; i < 8; ++i) { mu += red1[r][i]; m2 += red2[r][i]; }
    mu *= (1.f / COC);
    const float rstd = rsqrtf(m2 * (1.f / COC) - mu * mu + LNEPS);
#pragma unroll
    for (int jj = 0; jj < 32; ++jj) {
      const int c = li * 32 + jj;
      const float h = (sb[(size_t)r * DSTR + c] - mu) * rstd * a.gg[c] + a.gb[c];
      store_a_elem(Af, r, c, 8, (h16)fmaxf(h, 0.f));
    }
  }
  block_gemm_32(Af, a.Wg2h, a.bg2, sb, DSTR, 0, 8);
  __syncthreads();

  // ---- softmax over 32 neighbor rows, weighted sum of (v + pe) ----------
  {
    const int c = tid;
    const float scale = 0.17677669529663687f;  // 1/sqrt(HD=32)
    float mx = -3.4e38f;
#pragma unroll
    for (int rr = 0; rr < 32; ++rr) mx = fmaxf(mx, sb[(size_t)rr * DSTR + c] * scale);
    float e[32], ssum = 0.f;
#pragma unroll
    for (int rr = 0; rr < 32; ++rr) {
      e[rr] = __expf(sb[(size_t)rr * DSTR + c] * scale - mx);
      ssum += e[rr];
    }
    const float inv = 1.f / ssum;
    float acc = 0.f;
    for (int rr = 0; rr < 32; ++rr) {
      const float vv = a.v[(size_t)(b * NN + nb[rr]) * COC + c];  // coalesced over c
      acc += e[rr] * inv * (vv + pe[(size_t)rr * DSTR + c]);
    }
    a.out[(size_t)p * COC + c] = acc;
  }
}

// ---------------------------------------------------------------------------
// 6. fc_out GEMM + norm2
// ---------------------------------------------------------------------------
struct FcoutArgs {
  const float* ain; const h16* Wo; const float* bo;
  const float *g2, *b2n; float* out2;
};
__global__ void __launch_bounds__(256) fcout_kernel(FcoutArgs a) {
  __shared__ h16   Af[2 * 8 * FRAG_H];
  __shared__ float sb[32 * DSTR];
  __shared__ float red1[32][8], red2[32][8];
  const int row0 = blockIdx.x * 32, tid = threadIdx.x;
  for (int i = tid; i < 32 * COC; i += 256) {
    const int r = i >> 8, c = i & 255;
    store_a_elem(Af, r, c, 8, (h16)a.ain[(size_t)(row0 + r) * COC + c]);
  }
  block_gemm_32(Af, a.Wo, a.bo, sb, DSTR, 0, 8);
  __syncthreads();
  const int r = tid >> 3, li = tid & 7;
  float s = 0.f, s2 = 0.f;
#pragma unroll
  for (int jj = 0; jj < 32; ++jj) {
    const float v = sb[(size_t)r * DSTR + li * 32 + jj]; s += v; s2 += v * v;
  }
  red1[r][li] = s; red2[r][li] = s2; __syncthreads();
  float mu = 0.f, m2 = 0.f;
#pragma unroll
  for (int i = 0; i < 8; ++i) { mu += red1[r][i]; m2 += red2[r][i]; }
  mu *= (1.f / COC);
  const float rstd = rsqrtf(m2 * (1.f / COC) - mu * mu + LNEPS);
#pragma unroll
  for (int jj = 0; jj < 32; ++jj) {
    const int c = li * 32 + jj;
    a.out2[(size_t)(row0 + r) * COC + c] =
        (sb[(size_t)r * DSTR + c] - mu) * rstd * a.g2[c] + a.b2n[c];
  }
}

// ---------------------------------------------------------------------------
// 7. Channel attention: block per batch, thread per channel (coalesced mean).
// ---------------------------------------------------------------------------
struct CaArgs {
  const float* out2; const float *W1, *b1, *W2, *b2; float* cw;
};
__global__ void __launch_bounds__(256) chatt_kernel(CaArgs a) {
  __shared__ float mean[COC];
  __shared__ float hid[64];
  const int b = blockIdx.x, c = threadIdx.x;
  float s = 0.f;
  for (int n = 0; n < NN; ++n) s += a.out2[(size_t)(b * NN + n) * COC + c];
  mean[c] = s * (1.f / NN);
  __syncthreads();
  if (c < 64) {
    float h = a.b1[c];
    for (int i = 0; i < COC; ++i) h += mean[i] * a.W1[i * 64 + c];
    hid[c] = fmaxf(h, 0.f);
  }
  __syncthreads();
  float o = a.b2[c];
  for (int i = 0; i < 64; ++i) o += hid[i] * a.W2[i * COC + c];
  a.cw[(size_t)b * COC + c] = 1.f / (1.f + __expf(-o));
}

// ---------------------------------------------------------------------------
// 8. skip GEMM + norm3(out2*cw) + add
// ---------------------------------------------------------------------------
struct FinArgs {
  const h16* fth; const h16* Wsk; const float* bsk;
  const float* out2; const float* cw; const float *g3, *b3;
  float* out;
};
__global__ void __launch_bounds__(256) final_kernel(FinArgs a) {
  __shared__ h16   Af[2 * 4 * FRAG_H];  // Kc = 128
  __shared__ float sb[32 * DSTR];
  __shared__ float red1[32][8], red2[32][8];
  const int row0 = blockIdx.x * 32, tid = threadIdx.x;
  const int b = row0 / NN;  // 32 | NN so all rows share a batch
  for (int i = tid; i < 32 * CINC; i += 256) {
    const int r = i >> 7, c = i & 127;
    store_a_elem(Af, r, c, 4, a.fth[(size_t)(row0 + r) * CINC + c]);
  }
  block_gemm_32(Af, a.Wsk, a.bsk, sb, DSTR, 0, 4);
  __syncthreads();
  const int r = tid >> 3, li = tid & 7;
  float vb[32], s = 0.f, s2 = 0.f;
#pragma unroll
  for (int jj = 0; jj < 32; ++jj) {
    const int c = li * 32 + jj;
    const float v = a.out2[(size_t)(row0 + r) * COC + c] * a.cw[(size_t)b * COC + c];
    vb[jj] = v; s += v; s2 += v * v;
  }
  red1[r][li] = s; red2[r][li] = s2; __syncthreads();
  float mu = 0.f, m2 = 0.f;
#pragma unroll
  for (int i = 0; i < 8; ++i) { mu += red1[r][i]; m2 += red2[r][i]; }
  mu *= (1.f / COC);
  const float rstd = rsqrtf(m2 * (1.f / COC) - mu * mu + LNEPS);
#pragma unroll
  for (int jj = 0; jj < 32; ++jj) {
    const int c = li * 32 + jj;
    a.out[(size_t)(row0 + r) * COC + c] =
        (vb[jj] - mu) * rstd * a.g3[c] + a.b3[c] + sb[(size_t)r * DSTR + c];
  }
}

// ---------------------------------------------------------------------------
// Host launch. Input order assumption: setup_inputs() dict insertion order,
// params leaves flattened in insertion order (w,b / g,b per layer).
// ---------------------------------------------------------------------------
enum InIdx {
  IN_XYZ = 0, IN_FEAT,
  P_N1G, P_N1B,
  P_WQ, P_BQ, P_WK, P_BK, P_WV, P_BV,
  P_W1, P_B1, P_DLG, P_DLB, P_W2, P_B2,
  P_WG1, P_BG1, P_GLG, P_GLB, P_WG2, P_BG2,
  P_WO, P_BO,
  P_WCA1, P_BCA1, P_WCA2, P_BCA2,
  P_N2G, P_N2B, P_N3G, P_N3B,
  P_WSK, P_BSK
};

extern "C" void kernel_launch(void* const* d_in, const int* in_sizes, int n_in,
                              void* d_out, int out_size, void* d_ws, size_t ws_size,
                              hipStream_t stream) {
  (void)in_sizes; (void)n_in; (void)out_size; (void)ws_size;
  const float* xyz  = (const float*)d_in[IN_XYZ];
  const float* feat = (const float*)d_in[IN_FEAT];
  auto F = [&](int i) { return (const float*)d_in[i]; };

  // ---- workspace carve-up (256B aligned) ----
  char* base = (char*)d_ws;
  size_t off = 0;
  auto alloc = [&](size_t bytes) -> void* {
    off = (off + 255) & ~(size_t)255;
    void* p = base + off;
    off += bytes;
    return p;
  };
  h16*   fnh  = (h16*)  alloc((size_t)NPTS * CINC * sizeof(h16));
  h16*   fth  = (h16*)  alloc((size_t)NPTS * CINC * sizeof(h16));
  int*   idxb = (int*)  alloc((size_t)NPTS * KNB * sizeof(int));
  float* qb   = (float*)alloc((size_t)NPTS * COC * sizeof(float));
  float* kb   = (float*)alloc((size_t)NPTS * COC * sizeof(float));
  float* vb   = (float*)alloc((size_t)NPTS * COC * sizeof(float));
  float* attn = (float*)alloc((size_t)NPTS * COC * sizeof(float));
  float* out2 = (float*)alloc((size_t)NPTS * COC * sizeof(float));
  float* cw   = (float*)alloc((size_t)BB * COC * sizeof(float));
  h16* Wqh  = (h16*)alloc((size_t)CINC * COC * sizeof(h16));
  h16* Wkh  = (h16*)alloc((size_t)CINC * COC * sizeof(h16));
  h16* Wvh  = (h16*)alloc((size_t)CINC * COC * sizeof(h16));
  h16* W2h  = (h16*)alloc((size_t)COC * COC * sizeof(h16));
  h16* Wg1h = (h16*)alloc((size_t)COC * COC * sizeof(h16));
  h16* Wg2h = (h16*)alloc((size_t)COC * COC * sizeof(h16));
  h16* Woh  = (h16*)alloc((size_t)COC * COC * sizeof(h16));
  h16* Wskh = (h16*)alloc((size_t)CINC * COC * sizeof(h16));

  auto cvt = [&](int pidx, h16* dst, int n) {
    cvt_swz_b<<<(n + 255) / 256, 256, 0, stream>>>(F(pidx), dst, n);
  };
  cvt(P_WQ,  Wqh,  CINC * COC);
  cvt(P_WK,  Wkh,  CINC * COC);
  cvt(P_WV,  Wvh,  CINC * COC);
  cvt(P_W2,  W2h,  COC * COC);
  cvt(P_WG1, Wg1h, COC * COC);
  cvt(P_WG2, Wg2h, COC * COC);
  cvt(P_WO,  Woh,  COC * COC);
  cvt(P_WSK, Wskh, CINC * COC);

  ln1_kernel<<<NPTS, CINC, 0, stream>>>(feat, F(P_N1G), F(P_N1B), fnh, fth);
  knn_kernel<<<NPTS, 256, 0, stream>>>(xyz, idxb);

  QKVArgs qa{fnh, Wqh, Wkh, Wvh, F(P_BQ), F(P_BK), F(P_BV), qb, kb, vb};
  qkv_kernel<<<NPTS / 32, 256, 0, stream>>>(qa);

  AttnArgs aa{xyz, idxb, qb, kb, vb,
              F(P_W1), F(P_B1), F(P_DLG), F(P_DLB),
              W2h, F(P_B2),
              Wg1h, F(P_BG1), F(P_GLG), F(P_GLB),
              Wg2h, F(P_BG2),
              attn};
  attn_kernel<<<NPTS, 256, 0, stream>>>(aa);

  FcoutArgs fa{attn, Woh, F(P_BO), F(P_N2G), F(P_N2B), out2};
  fcout_kernel<<<NPTS / 32, 256, 0, stream>>>(fa);

  CaArgs ca{out2, F(P_WCA1), F(P_BCA1), F(P_WCA2), F(P_BCA2), cw};
  chatt_kernel<<<BB, 256, 0, stream>>>(ca);

  FinArgs fi{fth, Wskh, F(P_BSK), out2, cw, F(P_N3G), F(P_N3B), (float*)d_out};
  final_kernel<<<NPTS / 32, 256, 0, stream>>>(fi);
}
